// MetaLearner_24309514895364
// MI455X (gfx1250) — matrix operand order repaired and checked by reading, not compiled, not run on hardware
//
#include <hip/hip_runtime.h>

// ---------------- problem constants (match reference) ----------------
#define Hh   20          // hidden size
#define H4   80          // 4*H
#define NT   5           // N tiles of 16 covering 4H=80
#define LAY  2           // LSTM layers
#define EPSV 1e-5f
#define WPB  4           // waves per block (wave32)
#define TPB  (WPB * 32)  // threads per block
#define PRE_STRIDE 84    // multiple of 4 -> every row 16B aligned for b128 readback

typedef __attribute__((ext_vector_type(16))) _Float16     v16h;
typedef __attribute__((ext_vector_type(8)))  float        v8f;
typedef __attribute__((ext_vector_type(4)))  float        f32x4;
typedef __attribute__((ext_vector_type(4)))  unsigned int u32x4;

union AFrag { v16h h; u32x4 q[2]; };

static __device__ __forceinline__ unsigned int packh2(float a, float b) {
#if __has_builtin(__builtin_amdgcn_cvt_pkrtz)
  auto p = __builtin_amdgcn_cvt_pkrtz(a, b);     // single v_cvt_pk_f16_f32 (returns __fp16 x2)
  return __builtin_bit_cast(unsigned int, p);
#else
  unsigned short lo = __builtin_bit_cast(unsigned short, (_Float16)a);
  unsigned short hi = __builtin_bit_cast(unsigned short, (_Float16)b);
  return (unsigned int)lo | ((unsigned int)hi << 16);
#endif
}
static __device__ __forceinline__ float frcp(float a) {
#if __has_builtin(__builtin_amdgcn_rcpf)
  return __builtin_amdgcn_rcpf(a);
#else
  return 1.0f / a;
#endif
}
static __device__ __forceinline__ float frsq(float a) {
#if __has_builtin(__builtin_amdgcn_rsqf)
  return __builtin_amdgcn_rsqf(a);
#else
  return rsqrtf(a);
#endif
}
static __device__ __forceinline__ float fexp2(float a) {
#if __has_builtin(__builtin_amdgcn_exp2f)
  return __builtin_amdgcn_exp2f(a);
#else
  return exp2f(a);
#endif
}
static __device__ __forceinline__ float fsigmoid(float x) {
  return frcp(1.0f + fexp2(-1.4426950408889634f * x));
}
static __device__ __forceinline__ float ftanh_(float x) {
#if __has_builtin(__builtin_amdgcn_tanhf)
  return __builtin_amdgcn_tanhf(x);   // CDNA5 native V_TANH_F32
#else
  return 1.0f - 2.0f * frcp(1.0f + fexp2(2.8853900817779268f * x));
#endif
}

extern "C" __global__ __launch_bounds__(TPB)
void metalstm_kernel(const float* __restrict__ x,
                     const float* __restrict__ l1W,
                     const float* __restrict__ l1b,
                     const float* __restrict__ Wi,
                     const float* __restrict__ bi,
                     const float* __restrict__ bh,
                     const float* __restrict__ ln_i_g,
                     const float* __restrict__ ln_i_b,
                     const float* __restrict__ ln_h_g,
                     const float* __restrict__ ln_h_b,
                     const float* __restrict__ ln_c_g,
                     const float* __restrict__ ln_c_b,
                     const float* __restrict__ outW,
                     const float* __restrict__ outb,
                     float* __restrict__ out,
                     int B)
{
  // Block-shared, written once then read-only:
  __shared__ __attribute__((aligned(16))) unsigned short fragB[LAY][NT][32][16]; // Wi in WMMA B layout (f16)
  __shared__ float cBi[LAY][H4];   // bi (folded into the WMMA C-matrix)
  __shared__ float cGn[LAY][H4];   // ln_i_g
  __shared__ float cBc[LAY][H4];   // ln_i_b + LN(bh; ln_h_g, ln_h_b)  (hx==0 => Wh matmul vanishes)
  __shared__ float cCg[LAY][Hh], cCb[LAY][Hh];
  // Per-wave private staging (no barriers in hot loop; per-wave LDS ops are in-order):
  __shared__ __attribute__((aligned(16))) unsigned short aStage[WPB][32][32];    // f16 A tile, K padded to 32
  __shared__ __attribute__((aligned(16))) float preStage[WPB][32][PRE_STRIDE];   // biased WMMA accum, [row][80]

  const int tid  = threadIdx.x;
  const int lane = tid & 31;
  const int wv   = tid >> 5;

  // ---------- one-time block init ----------
  // Zero aStage once: K=24..31 halves stay zero forever (steady loop writes only K 0..23).
  for (int idx = tid; idx < WPB * 32 * 32; idx += TPB)
    ((unsigned short*)aStage)[idx] = 0;

  // Marshal Wi (f32 -> f16) into B-fragment lanes (32x16), this kernel's contiguous K order:
  //   lane n (0..15): N=n, K=0..15 ; lane n+16: N=n, K=16..31 (K>=20 padded 0)
  for (int s = tid; s < LAY * NT * 32; s += TPB) {
    int l  = s / (NT * 32);
    int t  = (s / 32) % NT;
    int ln = s & 31;
    int n  = 16 * t + (ln & 15);
    int k0 = (ln < 16) ? 0 : 16;
    unsigned short* dst = &fragB[l][t][ln][0];
    for (int j = 0; j < 16; ++j) {
      int k = k0 + j;
      float v = (k < Hh) ? Wi[(l * H4 + n) * Hh + k] : 0.0f;
      dst[j] = __builtin_bit_cast(unsigned short, (_Float16)v);
    }
  }
  // Fold the hx==0 branch: cBc = ln_i_b + LN(bh)*ln_h_g + ln_h_b  (constant per layer)
  for (int idx = tid; idx < LAY * H4; idx += TPB) {
    int l = idx / H4;
    float s0 = 0.f, s1 = 0.f;
    for (int j = 0; j < H4; ++j) { float b = bh[l * H4 + j]; s0 += b; s1 += b * b; }
    float mu  = s0 * (1.0f / H4);
    float inv = frsq(s1 * (1.0f / H4) - mu * mu + EPSV);
    ((float*)cBi)[idx] = bi[idx];
    ((float*)cGn)[idx] = ln_i_g[idx];
    ((float*)cBc)[idx] = ln_i_b[idx] + (bh[idx] - mu) * inv * ln_h_g[idx] + ln_h_b[idx];
  }
  for (int idx = tid; idx < LAY * Hh; idx += TPB) {
    ((float*)cCg)[idx] = ln_c_g[idx];
    ((float*)cCb)[idx] = ln_c_b[idx];
  }
  __syncthreads();

  // Uniform small weights -> regs, hoisted out of the loop
  float l1w[Hh], l1bv[Hh], ow[Hh];
#pragma unroll
  for (int h = 0; h < Hh; ++h) { l1w[h] = l1W[h]; l1bv[h] = l1b[h]; ow[h] = outW[h]; }
  const float ob = outb[0];

  const int nlo = lane & 15;
  const int k0h = (lane < 16) ? 0 : 16;       // A chunk origin: contiguous K order, same as fragB

  // Per-lane column biases for the WMMA C-matrix (loop-invariant): bias depends
  // only on the output column, and every element of a lane's C fragment shares
  // one column -> C = splat(bi[col]) computes A*B + bi inside the matrix pipe
  // (no post-WMMA VALU on D => no WMMA->VALU hazard NOPs).
  float bcol[LAY][NT];
#pragma unroll
  for (int l = 0; l < LAY; ++l)
#pragma unroll
    for (int t = 0; t < NT; ++t) bcol[l][t] = cBi[l][nlo + 16 * t];

  const int totalTiles = (B + 31) >> 5;       // 32 rows per wave tile
  const int waveG = blockIdx.x * WPB + wv;
  const int waveN = gridDim.x * WPB;

  for (int tile = waveG; tile < totalTiles; tile += waveN) {
    const int row = tile * 32 + lane;
    float xv = (row < B) ? x[row] : 0.0f;
    if (tile + waveN < totalTiles)
      __builtin_prefetch(&x[(tile + waveN) * 32 + lane], 0, 0);   // global_prefetch_b8

    float xt[Hh];
#pragma unroll
    for (int h = 0; h < Hh; ++h) xt[h] = xv * l1w[h] + l1bv[h];

#pragma unroll
    for (int l = 0; l < LAY; ++l) {
      // ---- marshal A: lane owns row `lane`; write K 0..23 halves (24..31 are pre-zeroed) ----
      unsigned int pk[12];
#pragma unroll
      for (int j = 0; j < 10; ++j) pk[j] = packh2(xt[2 * j], xt[2 * j + 1]);
      pk[10] = 0u; pk[11] = 0u;
      u32x4* wp = (u32x4*)&aStage[wv][lane][0];
      { u32x4 t0 = {pk[0], pk[1], pk[2],  pk[3] }; wp[0] = t0; }
      { u32x4 t1 = {pk[4], pk[5], pk[6],  pk[7] }; wp[1] = t1; }
      { u32x4 t2 = {pk[8], pk[9], pk[10], pk[11]}; wp[2] = t2; }

      // ---- WMMA: 2 M-tiles x 5 N-tiles, K=32 in one shot, bias via C; dump via LDS ----
#pragma unroll
      for (int mt = 0; mt < 2; ++mt) {
        AFrag A;
        const u32x4* ap = (const u32x4*)&aStage[wv][mt * 16 + nlo][k0h];
        A.q[0] = ap[0];   // K k0..k0+7
        A.q[1] = ap[1];   // K k0+8..k0+15
        const int rbase = mt * 16 + ((lane >> 4) << 3);   // C/D layout: lanes<16 row=r, lanes>=16 row=r+8
#pragma unroll
        for (int t = 0; t < NT; ++t) {
          AFrag Bf;
          const u32x4* bp = (const u32x4*)&fragB[l][t][lane][0];
          Bf.q[0] = bp[0];
          Bf.q[1] = bp[1];
          const float b = bcol[l][t];
          v8f cb = {b, b, b, b, b, b, b, b};
          v8f acc = __builtin_amdgcn_wmma_f32_16x16x32_f16(
                        false, A.h, false, Bf.h, (short)0, cb, false, false);
          const int col = nlo + 16 * t;
#pragma unroll
          for (int r = 0; r < 8; ++r) preStage[wv][rbase + r][col] = acc[r];
        }
      }

      // ---- lane-major LN + gates: lane owns row `lane`; pull row as 20x ds_load_b128 ----
      float pre[H4];
      {
        const f32x4* pv = (const f32x4*)&preStage[wv][lane][0];
#pragma unroll
        for (int j = 0; j < H4 / 4; ++j) {
          f32x4 v = pv[j];
          pre[4 * j + 0] = v.x; pre[4 * j + 1] = v.y;
          pre[4 * j + 2] = v.z; pre[4 * j + 3] = v.w;
        }
      }
      float s0 = 0.f, s1 = 0.f;
#pragma unroll
      for (int n = 0; n < H4; ++n) { s0 += pre[n]; s1 += pre[n] * pre[n]; }
      float mu  = s0 * (1.0f / H4);
      float inv = frsq(s1 * (1.0f / H4) - mu * mu + EPSV);

      float cxv[Hh], ovv[Hh];
      float cs0 = 0.f, cs1 = 0.f;
#pragma unroll
      for (int h = 0; h < Hh; ++h) {
        float vi = (pre[h]      - mu) * inv * cGn[l][h]      + cBc[l][h];
        float vo = (pre[40 + h] - mu) * inv * cGn[l][40 + h] + cBc[l][40 + h];
        float vg = (pre[60 + h] - mu) * inv * cGn[l][60 + h] + cBc[l][60 + h];
        float c  = fsigmoid(vi) * ftanh_(vg);   // cx = i*g  (forget gate dead: cx_prev == 0)
        cxv[h] = c; ovv[h] = vo;
        cs0 += c; cs1 += c * c;
      }
      float muc  = cs0 * (1.0f / Hh);
      float invc = frsq(cs1 * (1.0f / Hh) - muc * muc + EPSV);
#pragma unroll
      for (int h = 0; h < Hh; ++h) {
        float th = ftanh_((cxv[h] - muc) * invc * cCg[l][h] + cCb[l][h]);
        xt[h] = fsigmoid(ovv[h]) * th;   // hx = o * tanh(LN(cx)) -> input of next layer
      }
    }

    float o = ob;
#pragma unroll
    for (int h = 0; h < Hh; ++h) o += xt[h] * ow[h];
    if (row < B) out[row] = o;
  }
}

extern "C" void kernel_launch(void* const* d_in, const int* in_sizes, int n_in,
                              void* d_out, int out_size, void* d_ws, size_t ws_size,
                              hipStream_t stream) {
  (void)n_in; (void)out_size; (void)d_ws; (void)ws_size;
  const float* x     = (const float*)d_in[0];
  const float* l1W   = (const float*)d_in[1];
  const float* l1b   = (const float*)d_in[2];
  const float* Wi    = (const float*)d_in[3];
  const float* bi    = (const float*)d_in[4];
  /* d_in[5] = Wh is provably unused: hx==0 at every step */
  const float* bh    = (const float*)d_in[6];
  const float* lnig  = (const float*)d_in[7];
  const float* lnib  = (const float*)d_in[8];
  const float* lnhg  = (const float*)d_in[9];
  const float* lnhb  = (const float*)d_in[10];
  const float* lncg  = (const float*)d_in[11];
  const float* lncb  = (const float*)d_in[12];
  const float* outW  = (const float*)d_in[13];
  const float* outb  = (const float*)d_in[14];

  int B = in_sizes[0];                       // 64 * 20000 = 1,280,000
  int tiles  = (B + 31) / 32;                // 32 rows per wave
  int blocks = (tiles + WPB - 1) / WPB;
  if (blocks > 1280) blocks = 1280;          // persistent grid-stride waves
  if (blocks < 1)    blocks = 1;

  metalstm_kernel<<<blocks, TPB, 0, stream>>>(x, l1W, l1b, Wi, bi, bh,
                                              lnig, lnib, lnhg, lnhb, lncg, lncb,
                                              outW, outb, (float*)d_out, B);
}
